// MotifPredictor_16269336118078
// MI455X (gfx1250) — compile-verified, matrix operand order checked by reading, not run on hardware
//
#include <hip/hip_runtime.h>
#include <hip/hip_bf16.h>
#include <stdint.h>

// ---------------------------------------------------------------------------
// Problem constants (from reference setup_inputs)
// ---------------------------------------------------------------------------
#define NN 6144      // edge_ctx rows
#define HH 512       // hidden
#define EE 49152     // num union edges
#define UU 2048      // union feature dim
#define PP 4096      // pooling dim
#define RR 51        // relation classes
#define NUM_OBJ 151

#define AS1 __attribute__((address_space(1)))
#define AS3 __attribute__((address_space(3)))

// Optional CDNA5 async global->LDS copy path (ASYNCcnt-tracked), guarded so
// the kernel still compiles on toolchains without the builtins.
#if defined(__has_builtin)
#  if __has_builtin(__builtin_amdgcn_global_load_async_to_lds_b128) && \
      __has_builtin(__builtin_amdgcn_s_wait_asynccnt)
#    define USE_ASYNC_LDS 1
#  endif
#endif
#ifndef USE_ASYNC_LDS
#  define USE_ASYNC_LDS 0
#endif

// ---------------------------------------------------------------------------
// bf16 WMMA plumbing (CDNA5: v_wmma_f32_16x16x32_bf16, fp32 accumulate)
// ---------------------------------------------------------------------------
typedef __attribute__((ext_vector_type(16))) __bf16 v16bf;
typedef __attribute__((ext_vector_type(8)))  float  v8f;
typedef __attribute__((ext_vector_type(4)))  int    v4i;

union Frag { unsigned int u[8]; v16bf v; };

#define WMMA_BF16(A, B, C) \
  __builtin_amdgcn_wmma_f32_16x16x32_bf16(false, (A), false, (B), (short)0, (C), false, false)

__device__ __forceinline__ unsigned short f2bf(float f) {
  unsigned int x = __float_as_uint(f);
  x += 0x7FFFu + ((x >> 16) & 1u);   // round-to-nearest-even
  return (unsigned short)(x >> 16);
}
__device__ __forceinline__ unsigned int pack2bf(float lo, float hi) {
  return (unsigned int)f2bf(lo) | ((unsigned int)f2bf(hi) << 16);
}

// K-pair offset within a bf16 A/B fragment per ISA layout:
// vgpr v (v<4 -> K block 0..15, v>=4 -> 16..31), lane half h adds K+8.
__device__ __forceinline__ int kpOff(int v, int h) {
  return ((v & 4) << 1) + h * 4 + (v & 3);
}

__device__ __forceinline__ Frag loadFragA(const unsigned int (*lA)[17], int row0, int lane) {
  Frag f; const int r = row0 + (lane & 15); const int h = lane >> 4;
#pragma unroll
  for (int v = 0; v < 8; ++v) f.u[v] = lA[r][kpOff(v, h)];
  return f;
}
__device__ __forceinline__ Frag loadFragB(const unsigned int (*lB)[17], int n0, int lane) {
  Frag f; const int n = n0 + (lane & 15); const int h = lane >> 4;
#pragma unroll
  for (int v = 0; v < 8; ++v) f.u[v] = lB[n][kpOff(v, h)];
  return f;
}

// ---------------------------------------------------------------------------
// Tile staging: global -> VGPR (with f32->bf16 pack) and VGPR -> LDS commit
// ---------------------------------------------------------------------------
struct Stage8 { unsigned int u[8]; };

// 128x32 fp32 A tile: thread t handles row t>>1, 16-wide K half t&1
__device__ __forceinline__ Stage8 stageA_f32(const float* __restrict__ src,
                                             long row0, int k0, int ld, int t) {
  const int m = t >> 1, half = t & 1;
  const float4* p = (const float4*)(src + (row0 + m) * (long)ld + k0 + half * 16);
  float4 q0 = p[0], q1 = p[1], q2 = p[2], q3 = p[3];
  Stage8 s;
  s.u[0] = pack2bf(q0.x, q0.y); s.u[1] = pack2bf(q0.z, q0.w);
  s.u[2] = pack2bf(q1.x, q1.y); s.u[3] = pack2bf(q1.z, q1.w);
  s.u[4] = pack2bf(q2.x, q2.y); s.u[5] = pack2bf(q2.z, q2.w);
  s.u[6] = pack2bf(q3.x, q3.y); s.u[7] = pack2bf(q3.z, q3.w);
  return s;
}

// 128x32 bf16 A tile (already bf16 in memory): pure 32-byte copy per thread
__device__ __forceinline__ Stage8 stageA_bf16(const unsigned short* __restrict__ src,
                                              long row, int k0, int t) {
  const int half = t & 1;
  const uint4* p = (const uint4*)(src + row * (2 * HH) + k0 + half * 16);
  uint4 q0 = p[0], q1 = p[1];
  Stage8 s;
  s.u[0] = q0.x; s.u[1] = q0.y; s.u[2] = q0.z; s.u[3] = q0.w;
  s.u[4] = q1.x; s.u[5] = q1.y; s.u[6] = q1.z; s.u[7] = q1.w;
  return s;
}

__device__ __forceinline__ void storeTileA(unsigned int (*lA)[17], const Stage8& s, int t) {
  const int m = t >> 1, half = t & 1;
  unsigned int* d = &lA[m][half * 8];
#pragma unroll
  for (int j = 0; j < 8; ++j) d[j] = s.u[j];
}

// 32x128 fp32 B tile (row-major, leading dim ld): thread t covers 4 cols x 2 K-pairs
__device__ __forceinline__ Stage8 stageB_f32(const float* __restrict__ src,
                                             int k0, int n0, int ld, int t) {
  const int n4 = (t & 31) * 4;
  const int kp0 = t >> 5;
  Stage8 s;
#pragma unroll
  for (int it = 0; it < 2; ++it) {
    const int kp = kp0 + it * 8;
    const float* r0 = src + (long)(k0 + 2 * kp) * ld + n0 + n4;
    float4 a = *(const float4*)r0;
    float4 b = *(const float4*)(r0 + ld);
    s.u[it * 4 + 0] = pack2bf(a.x, b.x);
    s.u[it * 4 + 1] = pack2bf(a.y, b.y);
    s.u[it * 4 + 2] = pack2bf(a.z, b.z);
    s.u[it * 4 + 3] = pack2bf(a.w, b.w);
  }
  return s;
}
__device__ __forceinline__ void storeTileB(unsigned int (*lB)[17], const Stage8& s, int t) {
  const int n4 = (t & 31) * 4;
  const int kp0 = t >> 5;
#pragma unroll
  for (int it = 0; it < 2; ++it) {
    const int kp = kp0 + it * 8;
#pragma unroll
    for (int j = 0; j < 4; ++j) lB[n4 + j][kp] = s.u[it * 4 + j];
  }
}

#if USE_ASYNC_LDS
// Async 16B global->LDS copy (no VGPR staging, tracked by ASYNCcnt).
// Builtin signature (from hipcc diagnostic): (v4i AS1* src, v4i AS3* dst, imm, imm)
__device__ __forceinline__ void asyncCopy16(const void* gsrc, void* ldst) {
  __builtin_amdgcn_global_load_async_to_lds_b128((AS1 v4i*)gsrc, (AS3 v4i*)ldst, 0, 0);
}
// issue the 32B/thread A tile copy for one thread
__device__ __forceinline__ void asyncTileA_bf16(unsigned int (*lA)[17],
                                                const unsigned short* __restrict__ src,
                                                long row, int k0, int t) {
  const int m = t >> 1, half = t & 1;
  const unsigned short* g = src + row * (2 * HH) + k0 + half * 16;
  void* l = (void*)&lA[m][half * 8];
  asyncCopy16(g, l);
  asyncCopy16(g + 8, (char*)l + 16);
}
#endif

// 8 WMMAs on the current LDS tiles (wave = 32 rows x 64 cols of the 128x128 tile)
__device__ __forceinline__ void mainCompute(const unsigned int (*lA)[17],
                                            const unsigned int (*lB)[17],
                                            v8f (&acc)[2][4], int wm, int wn, int lane) {
  Frag a[2], b[4];
#pragma unroll
  for (int sm = 0; sm < 2; ++sm) a[sm] = loadFragA(lA, wm * 32 + sm * 16, lane);
#pragma unroll
  for (int sn = 0; sn < 4; ++sn) b[sn] = loadFragB(lB, wn * 64 + sn * 16, lane);
#pragma unroll
  for (int sm = 0; sm < 2; ++sm)
#pragma unroll
    for (int sn = 0; sn < 4; ++sn)
      acc[sm][sn] = WMMA_BF16(a[sm].v, b[sn].v, acc[sm][sn]);
}

// ---------------------------------------------------------------------------
// Kernel 1: edge_rep = edge_ctx @ W_post_emb + b_post_emb   -> bf16 workspace
// Grid (48, 8), block 256 (8 waves; 4x2 wave tiling of a 128x128 tile)
// ---------------------------------------------------------------------------
__global__ __launch_bounds__(256)
void k_edge_rep(const float* __restrict__ edge_ctx, const float* __restrict__ W,
                const float* __restrict__ bias, unsigned short* __restrict__ edge_rep) {
  __shared__ unsigned int lA[128][17];
  __shared__ unsigned int lB[128][17];
  const int t = threadIdx.x, lane = t & 31, wid = t >> 5;
  const int wm = wid >> 1, wn = wid & 1;
  const long rb = (long)blockIdx.x * 128;
  const int  cb = blockIdx.y * 128;

  v8f acc[2][4] = {};
  for (int k0 = 0; k0 < HH; k0 += 32) {
    __syncthreads();
    Stage8 sa = stageA_f32(edge_ctx, rb, k0, HH, t);
    Stage8 sb = stageB_f32(W, k0, cb, 2 * HH, t);
    storeTileA(lA, sa, t);
    storeTileB(lB, sb, t);
    __syncthreads();
    mainCompute(lA, lB, acc, wm, wn, lane);
  }
  const int h = lane >> 4, nl = lane & 15;
#pragma unroll
  for (int sm = 0; sm < 2; ++sm)
#pragma unroll
    for (int sn = 0; sn < 4; ++sn) {
      const int col = cb + wn * 64 + sn * 16 + nl;
      const float bc = bias[col];
#pragma unroll
      for (int v = 0; v < 8; ++v) {
        const long row = rb + wm * 32 + sm * 16 + h * 8 + v;
        edge_rep[row * (2 * HH) + col] = f2bf(acc[sm][sn][v] + bc);
      }
    }
}

// ---------------------------------------------------------------------------
// Kernel 2: out[e][r] = b_rel[r] + freq_table[obj[p0]*151 + obj[p1]][r]
// ---------------------------------------------------------------------------
__global__ void k_init_out(const float* __restrict__ b_rel, const float* __restrict__ freq,
                           const int* __restrict__ obj, const int* __restrict__ pairs,
                           float* __restrict__ out) {
  const long i = (long)blockIdx.x * 256 + threadIdx.x;
  if (i >= (long)EE * RR) return;
  const long e = i / RR;
  const int  r = (int)(i - e * RR);
  const int p0 = pairs[e * 2 + 0];
  const int p1 = pairs[e * 2 + 1];
  const long bi = (long)obj[p0] * NUM_OBJ + obj[p1];
  out[i] = b_rel[r] + freq[bi * RR + r];
}

// ---------------------------------------------------------------------------
// Kernel 3 (fused, double-buffered): per 128-edge x 128-P tile:
//   t1 = gather(edge_rep) @ W_post_cat + b_post_cat     (K=1024, bf16 WMMA,
//                                                        A tile via async->LDS)
//   t2 = union_features   @ W_up       + b_up           (K=2048, bf16 WMMA)
//   prod = t1 * t2  -> bf16 in LDS
//   out[tile rows] += prod @ W_rel[pb:pb+128, :]        (K=128, bf16 WMMA,
//                                                        fp32 global atomics)
// Grid (384, 32), block 256.
// ---------------------------------------------------------------------------
__global__ __launch_bounds__(256)
void k_fused(const unsigned short* __restrict__ edge_rep,
             const float* __restrict__ unionf,
             const float* __restrict__ W_cat, const float* __restrict__ b_cat,
             const float* __restrict__ W_up,  const float* __restrict__ b_up,
             const float* __restrict__ W_rel, const int* __restrict__ pairs,
             float* __restrict__ out) {
  __shared__ union {
    struct { unsigned int lA[2][128][17]; unsigned int lB[2][128][17]; } g;  // 34816 B
    struct { unsigned short prod[128][132]; unsigned int wr[64][65]; } r;    // 50432 B
  } sh;

  const int t = threadIdx.x, lane = t & 31, wid = t >> 5;
  const int wm = wid >> 1, wn = wid & 1;
  const int h = lane >> 4, nl = lane & 15;
  const long eb = (long)blockIdx.x * 128;
  const int  pb = blockIdx.y * 128;

  // gather indices for the A-loader thread's edge row (thread t loads row t>>1)
  const int am = t >> 1;
  const long my_e = eb + am;
  const int p0 = pairs[my_e * 2 + 0];
  const int p1 = pairs[my_e * 2 + 1];

  // ---- GEMM2: gathered edge_rep (bf16) @ W_post_cat, K = 1024 ----
  v8f acc1[2][4] = {};
  {
    // prologue: fill buffer 0 with k0 = 0 (head half -> p0)
#if USE_ASYNC_LDS
    asyncTileA_bf16(sh.g.lA[0], edge_rep, p0, 0, t);
#else
    Stage8 sa0 = stageA_bf16(edge_rep, p0, 0, t);
    storeTileA(sh.g.lA[0], sa0, t);
#endif
    Stage8 sb0 = stageB_f32(W_cat, 0, pb, PP, t);
    storeTileB(sh.g.lB[0], sb0, t);
#if USE_ASYNC_LDS
    __builtin_amdgcn_s_wait_asynccnt(0);
#endif
    __syncthreads();

    for (int k0 = 0; k0 < 2 * HH; k0 += 32) {
      const int cur = (k0 >> 5) & 1, nxt = cur ^ 1;
      const bool more = (k0 + 32) < 2 * HH;
      Stage8 sbN;
#if !USE_ASYNC_LDS
      Stage8 saN;
#endif
      if (more) {
        // each 32-wide K tile lies entirely in head (k<512 -> p0) or tail (p1)
        const int prow = (k0 + 32 < HH) ? p0 : p1;
#if USE_ASYNC_LDS
        asyncTileA_bf16(sh.g.lA[nxt], edge_rep, prow, k0 + 32, t);
#else
        saN = stageA_bf16(edge_rep, prow, k0 + 32, t);
#endif
        sbN = stageB_f32(W_cat, k0 + 32, pb, PP, t);
      }
      mainCompute(sh.g.lA[cur], sh.g.lB[cur], acc1, wm, wn, lane);
      if (more) {
#if !USE_ASYNC_LDS
        storeTileA(sh.g.lA[nxt], saN, t);
#endif
        storeTileB(sh.g.lB[nxt], sbN, t);
      }
#if USE_ASYNC_LDS
      __builtin_amdgcn_s_wait_asynccnt(0);
#endif
      __syncthreads();
    }
  }

  // ---- GEMM3: union_features (fp32 -> bf16) @ W_up, K = 2048 ----
  v8f acc2[2][4] = {};
  {
    Stage8 sa0 = stageA_f32(unionf, eb, 0, UU, t);
    Stage8 sb0 = stageB_f32(W_up, 0, pb, PP, t);
    storeTileA(sh.g.lA[0], sa0, t);
    storeTileB(sh.g.lB[0], sb0, t);
    __syncthreads();

    for (int k0 = 0; k0 < UU; k0 += 32) {
      const int cur = (k0 >> 5) & 1, nxt = cur ^ 1;
      const bool more = (k0 + 32) < UU;
      Stage8 saN, sbN;
      if (more) {
        saN = stageA_f32(unionf, eb, k0 + 32, UU, t);
        sbN = stageB_f32(W_up, k0 + 32, pb, PP, t);
      }
      mainCompute(sh.g.lA[cur], sh.g.lB[cur], acc2, wm, wn, lane);
      if (more) {
        storeTileA(sh.g.lA[nxt], saN, t);
        storeTileB(sh.g.lB[nxt], sbN, t);
      }
      __syncthreads();
    }
  }

  // GEMM tiles (sh.g) are dead past this point (loop ends with a barrier);
  // sh.r reuses that LDS.

  // ---- elementwise: prod = (t1 + b_cat) * (t2 + b_up) -> bf16 in LDS ----
#pragma unroll
  for (int sm = 0; sm < 2; ++sm)
#pragma unroll
    for (int sn = 0; sn < 4; ++sn) {
      const int lcol = wn * 64 + sn * 16 + nl;
      const int col  = pb + lcol;
      const float bc = b_cat[col], bu = b_up[col];
#pragma unroll
      for (int v = 0; v < 8; ++v) {
        const int lrow = wm * 32 + sm * 16 + h * 8 + v;
        const float p = (acc1[sm][sn][v] + bc) * (acc2[sm][sn][v] + bu);
        sh.r.prod[lrow][lcol] = f2bf(p);
      }
    }

  // ---- stage W_rel[pb:pb+128, 0:51] as bf16 K-pairs (cols padded to 64) ----
  {
    const int n = t & 63;
    const int kpBase = (t >> 6) * 16;
#pragma unroll
    for (int i = 0; i < 16; ++i) {
      const int kp = kpBase + i;
      unsigned int val = 0;
      if (n < RR) {
        const float lo = W_rel[(long)(pb + 2 * kp + 0) * RR + n];
        const float hi = W_rel[(long)(pb + 2 * kp + 1) * RR + n];
        val = pack2bf(lo, hi);
      }
      sh.r.wr[n][kp] = val;
    }
  }
  __syncthreads();

  // ---- GEMM4: prod(128x128 bf16) @ W_rel slice (128x64), split-K atomics ----
  {
    const int mb = wid * 16;  // each wave: 16 rows x 64 cols, K = 128
    v8f racc[4] = {};
#pragma unroll
    for (int ks = 0; ks < 4; ++ks) {
      Frag a;
      {
        const int r = mb + nl;
        const unsigned int* pr = (const unsigned int*)&sh.r.prod[r][0];
#pragma unroll
        for (int v = 0; v < 8; ++v) a.u[v] = pr[ks * 16 + kpOff(v, h)];
      }
#pragma unroll
      for (int sn = 0; sn < 4; ++sn) {
        Frag b;
#pragma unroll
        for (int v = 0; v < 8; ++v) b.u[v] = sh.r.wr[sn * 16 + nl][ks * 16 + kpOff(v, h)];
        racc[sn] = WMMA_BF16(a.v, b.v, racc[sn]);
      }
    }
#pragma unroll
    for (int sn = 0; sn < 4; ++sn) {
      const int n = sn * 16 + nl;
      if (n < RR) {
#pragma unroll
        for (int v = 0; v < 8; ++v) {
          const long row = eb + mb + h * 8 + v;
          atomicAdd(&out[row * RR + n], racc[sn][v]);
        }
      }
    }
  }
}

// ---------------------------------------------------------------------------
// Launch
// ---------------------------------------------------------------------------
extern "C" void kernel_launch(void* const* d_in, const int* in_sizes, int n_in,
                              void* d_out, int out_size, void* d_ws, size_t ws_size,
                              hipStream_t stream) {
  const float* edge_ctx   = (const float*)d_in[0];
  const float* unionf     = (const float*)d_in[1];
  const float* W_post_emb = (const float*)d_in[2];
  const float* b_post_emb = (const float*)d_in[3];
  const float* W_post_cat = (const float*)d_in[4];
  const float* b_post_cat = (const float*)d_in[5];
  const float* W_up       = (const float*)d_in[6];
  const float* b_up       = (const float*)d_in[7];
  const float* W_rel      = (const float*)d_in[8];
  const float* b_rel      = (const float*)d_in[9];
  const float* freq       = (const float*)d_in[10];
  const int*   obj        = (const int*)d_in[11];
  const int*   pairs      = (const int*)d_in[12];
  float* out = (float*)d_out;
  unsigned short* edge_rep = (unsigned short*)d_ws;  // NN x 1024 bf16 = 12.6 MB

  (void)in_sizes; (void)n_in; (void)out_size; (void)ws_size;

  k_edge_rep<<<dim3(NN / 128, (2 * HH) / 128), 256, 0, stream>>>(
      edge_ctx, W_post_emb, b_post_emb, edge_rep);

  const long tot = (long)EE * RR;
  k_init_out<<<(unsigned)((tot + 255) / 256), 256, 0, stream>>>(
      b_rel, freq, obj, pairs, out);

  k_fused<<<dim3(EE / 128, PP / 128), 256, 0, stream>>>(
      edge_rep, unionf, W_post_cat, b_post_cat, W_up, b_up, W_rel, pairs, out);
}
